// CorticalColumnGroup_82867099009521
// MI455X (gfx1250) — matrix operand order repaired and checked by reading, not compiled, not run on hardware
//
#include <hip/hip_runtime.h>
#include <hip/hip_bf16.h>

#define BSZ 4
#define SEQ 512
#define GRP 64
#define DIM 128
#define HID 512
#define DPD 64

typedef __bf16 bf16_t;
typedef __attribute__((ext_vector_type(16))) __bf16 v16bf;
typedef __attribute__((ext_vector_type(8)))  float  v8f;
typedef int v4i_t __attribute__((vector_size(16)));

// Async memory->LDS 16-byte copy (CDNA5 GLOBAL_LOAD_ASYNC_TO_LDS_B128),
// with a synchronous fallback when the builtin is unavailable.
static __device__ __forceinline__ void async_copy16(void* ldst, const void* gsrc) {
#if __has_builtin(__builtin_amdgcn_global_load_async_to_lds_b128)
  __builtin_amdgcn_global_load_async_to_lds_b128((v4i_t*)gsrc, (v4i_t*)ldst, 0, 0);
#else
  *(float4*)ldst = *(const float4*)gsrc;
#endif
}

static __device__ __forceinline__ void async_wait() {
#if __has_builtin(__builtin_amdgcn_s_wait_asynccnt)
  __builtin_amdgcn_s_wait_asynccnt(0);
#else
  asm volatile("s_wait_asynccnt 0" ::: "memory");
#endif
}

static __device__ __forceinline__ v8f wmma_bf16(v16bf a, v16bf b, v8f c) {
  // D = A(16x32 bf16) * B(32x16 bf16) + C(16x16 f32)
  return __builtin_amdgcn_wmma_f32_16x16x32_bf16(false, a, false, b, (short)0, c, false, false);
}

// A fragment: 16x32 slice (cols kbase..kbase+31) of LDS row-major bf16 matrix.
// Lanes 0-15 hold M=lane, K 0..15; lanes 16-31 hold K 16..31 (matches load_b).
static __device__ __forceinline__ v16bf load_a(const bf16_t* s, int lda, int lane, int kbase) {
  const int half = lane >> 4, m = lane & 15;
  const bf16_t* row = s + m * lda + kbase + 16 * half;
  v16bf a;
#pragma unroll
  for (int j = 0; j < 8; ++j) {
    const int k0 = 2 * j;
    a[2 * j]     = row[k0];
    a[2 * j + 1] = row[k0 + 1];
  }
  return a;
}

// B fragment: 32x16 slice; source is transposed weight wt[N][K] row-major.
// Lanes 0-15 hold K=0..15 of column n=lane; lanes 16-31 hold K=16..31.
static __device__ __forceinline__ v16bf load_b(const bf16_t* wt, int ldk, int lane, int kbase) {
  const int half = lane >> 4, n = lane & 15;
  const bf16_t* row = wt + n * ldk + kbase + 16 * half;
  v16bf b;
#pragma unroll
  for (int j = 0; j < 8; ++j) {
    b[2 * j]     = row[2 * j];
    b[2 * j + 1] = row[2 * j + 1];
  }
  return b;
}

template <int KTOT>
static __device__ __forceinline__ v8f gemm16(const bf16_t* aLds, int lda,
                                             const bf16_t* wt, int ldk,
                                             int lane, v8f acc) {
#pragma unroll
  for (int kk = 0; kk < KTOT / 32; ++kk) {
    v16bf a = load_a(aLds, lda, lane, kk * 32);
    v16bf b = load_b(wt, ldk, lane, kk * 32);
    acc = wmma_bf16(a, b, acc);
  }
  return acc;
}

// ---------------------------------------------------------------------------
// Weight prep: dst[g][n][k] = (bf16) src[g][k][n]
__global__ void k_transpose_bf16(const float* __restrict__ src, bf16_t* __restrict__ dst,
                                 int G, int K, int N) {
  long long idx = (long long)blockIdx.x * blockDim.x + threadIdx.x;
  long long total = (long long)G * K * N;
  if (idx >= total) return;
  int n = (int)(idx % N);
  long long t = idx / N;
  int k = (int)(t % K);
  int g = (int)(t / K);
  dst[((long long)g * N + n) * K + k] = (bf16_t)src[idx];
}

// ---------------------------------------------------------------------------
// Stage 1: LN + attention + residual. One wave handles 16 rows (2 attention
// blocks of 8) of the flattened (bs*n*G, D) matrix.
__global__ __launch_bounds__(32)
void k_attention(const float* __restrict__ x,
                 const float* __restrict__ lat_g, const float* __restrict__ lat_b,
                 const bf16_t* __restrict__ wq_t, const bf16_t* __restrict__ wk_t,
                 const bf16_t* __restrict__ wv_t, const bf16_t* __restrict__ wo_t,
                 const float* __restrict__ bq, const float* __restrict__ bk,
                 const float* __restrict__ bv, const float* __restrict__ bo,
                 float* __restrict__ x_mid) {
  __shared__ float  xs[16 * DIM];
  __shared__ bf16_t hs[16 * DIM];
  __shared__ float  qs[16 * DIM];
  __shared__ float  ks2[16 * DIM];
  __shared__ float  vs[16 * DIM];
  __shared__ bf16_t os[16 * DIM];
  __shared__ float  att[2 * 8 * 8];
  __shared__ float  red[64];

  const int lane = threadIdx.x;
  const long long rbase = (long long)blockIdx.x * 16;
  const float* xin = x + rbase * DIM;

  // async-stage the 16x128 f32 tile into LDS (16B chunks, 16 per lane)
#pragma unroll
  for (int c = 0; c < 16; ++c) {
    const int i = (c * 32 + lane) * 4; // float index, 16B aligned
    async_copy16(xs + i, xin + i);
  }
  async_wait();
  __syncthreads();

  { // LayerNorm -> hs
    const int row = lane >> 1, seg = lane & 1;
    const float* xr = xs + row * DIM + seg * 64;
    float s = 0.f, s2 = 0.f;
#pragma unroll
    for (int t = 0; t < 64; ++t) { float v = xr[t]; s += v; s2 += v * v; }
    red[lane] = s; red[32 + lane] = s2;
    __syncthreads();
    const float sum = red[row * 2] + red[row * 2 + 1];
    const float sq  = red[32 + row * 2] + red[32 + row * 2 + 1];
    const float mean = sum * (1.f / DIM);
    const float var  = sq * (1.f / DIM) - mean * mean;
    const float rstd = rsqrtf(var + 1e-5f);
    const float* gg = lat_g + seg * 64;
    const float* bb = lat_b + seg * 64;
    bf16_t* hr = hs + row * DIM + seg * 64;
#pragma unroll
    for (int t = 0; t < 64; ++t)
      hr[t] = (bf16_t)((xr[t] - mean) * rstd * gg[t] + bb[t]);
  }
  __syncthreads();

  const int half = lane >> 4, mn = lane & 15;
#pragma unroll 1
  for (int nt = 0; nt < 8; ++nt) { // q,k,v projections
    if (nt + 1 < 8) {
      __builtin_prefetch(wq_t + (nt + 1) * 16 * DIM, 0, 1);
      __builtin_prefetch(wk_t + (nt + 1) * 16 * DIM, 0, 1);
      __builtin_prefetch(wv_t + (nt + 1) * 16 * DIM, 0, 1);
    }
    const int c = nt * 16 + mn;
    v8f aq, ak, av;
    const float biq = bq[c], bik = bk[c], biv = bv[c];
#pragma unroll
    for (int r = 0; r < 8; ++r) { aq[r] = biq; ak[r] = bik; av[r] = biv; }
    aq = gemm16<DIM>(hs, DIM, wq_t + nt * 16 * DIM, DIM, lane, aq);
    ak = gemm16<DIM>(hs, DIM, wk_t + nt * 16 * DIM, DIM, lane, ak);
    av = gemm16<DIM>(hs, DIM, wv_t + nt * 16 * DIM, DIM, lane, av);
#pragma unroll
    for (int r = 0; r < 8; ++r) {
      const int row = half * 8 + r;
      qs[row * DIM + c]  = aq[r];
      ks2[row * DIM + c] = ak[r];
      vs[row * DIM + c]  = av[r];
    }
  }
  __syncthreads();

  for (int t = lane; t < 128; t += 32) { // scores
    const int blk = t >> 6, rc = t & 63, rr = rc >> 3, cc = rc & 7;
    const float* qr = qs + (blk * 8 + rr) * DIM;
    const float* kr = ks2 + (blk * 8 + cc) * DIM;
    float s = 0.f;
#pragma unroll
    for (int i = 0; i < DIM; ++i) s += qr[i] * kr[i];
    att[t] = s * 0.08838834764831845f; // 1/sqrt(128)
  }
  __syncthreads();

  if (lane < 16) { // softmax per (blk,row) over 8 entries
    float* ar = att + (lane >> 3) * 64 + (lane & 7) * 8;
    float mx = ar[0];
#pragma unroll
    for (int e = 1; e < 8; ++e) mx = fmaxf(mx, ar[e]);
    float sm = 0.f;
#pragma unroll
    for (int e = 0; e < 8; ++e) { float v = __expf(ar[e] - mx); ar[e] = v; sm += v; }
    const float inv = 1.f / sm;
#pragma unroll
    for (int e = 0; e < 8; ++e) ar[e] *= inv;
  }
  __syncthreads();

  for (int i = lane; i < 16 * DIM; i += 32) { // o = attn @ v
    const int row = i >> 7, col = i & 127, blk = row >> 3;
    const float* ar = att + blk * 64 + (row & 7) * 8;
    const float* vr = vs + blk * 8 * DIM + col;
    float s = 0.f;
#pragma unroll
    for (int e = 0; e < 8; ++e) s += ar[e] * vr[e * DIM];
    os[i] = (bf16_t)s;
  }
  __syncthreads();

  float* xout = x_mid + rbase * DIM;
#pragma unroll 1
  for (int nt = 0; nt < 8; ++nt) { // x_mid = x + o @ wo + bo
    if (nt + 1 < 8) __builtin_prefetch(wo_t + (nt + 1) * 16 * DIM, 0, 1);
    const int c = nt * 16 + mn;
    v8f acc;
    const float bi = bo[c];
#pragma unroll
    for (int r = 0; r < 8; ++r) acc[r] = bi;
    acc = gemm16<DIM>(os, DIM, wo_t + nt * 16 * DIM, DIM, lane, acc);
#pragma unroll
    for (int r = 0; r < 8; ++r) {
      const int row = half * 8 + r;
      xout[row * DIM + c] = xs[row * DIM + c] + acc[r];
    }
  }
}

// ---------------------------------------------------------------------------
// Stage 2: grouped enc/surprise/gain/pred/gln/MLP. Grid = G * (T/16); one wave
// handles 16 tokens of one group (rows strided by GRP*DIM).
__global__ __launch_bounds__(32)
void k_group_stage(const float* __restrict__ x_mid,
                   const float* __restrict__ z_hat_prev,
                   const bf16_t* __restrict__ enc_t, const float* __restrict__ enc_b,
                   const bf16_t* __restrict__ gain_t, const float* __restrict__ gain_b,
                   const bf16_t* __restrict__ pred_t, const float* __restrict__ pred_b,
                   const float* __restrict__ ln_g, const float* __restrict__ ln_b,
                   const bf16_t* __restrict__ up_t, const float* __restrict__ up_b,
                   const bf16_t* __restrict__ down_t, const float* __restrict__ down_b,
                   float* __restrict__ out_z, float* __restrict__ out_zhat,
                   float* __restrict__ out_sur, float* __restrict__ out_gate,
                   float* __restrict__ out_x) {
  __shared__ float  xres[16 * DIM];
  __shared__ bf16_t ax[16 * DIM];
  __shared__ float  dfull[16 * DPD];
  __shared__ bf16_t adz[16 * DPD];
  __shared__ bf16_t az[16 * DPD];
  __shared__ float  gains[16 * DIM];
  __shared__ bf16_t ah[16 * DIM];
  __shared__ bf16_t au[16 * HID];
  __shared__ float  red[64];

  const int lane = threadIdx.x;
  const int g  = blockIdx.x >> 7;
  const int tt = blockIdx.x & 127;
  const int half = lane >> 4, mn = lane & 15;

  // async-stage 16 rows (each row 512B contiguous, global stride GRP*DIM)
#pragma unroll
  for (int c = 0; c < 16; ++c) {
    const int chunk = c * 32 + lane;        // 0..511 16B chunks
    const int row = chunk >> 5, col4 = chunk & 31;
    const long long fr = (long long)(tt * 16 + row) * GRP + g;
    async_copy16(xres + chunk * 4, x_mid + fr * DIM + col4 * 4);
  }
  async_wait();
  __syncthreads();

  for (int i = lane; i < 16 * DIM; i += 32) ax[i] = (bf16_t)xres[i];
  __syncthreads();

  const bf16_t* encw = enc_t + (long long)g * DPD * DIM;
#pragma unroll 1
  for (int nt = 0; nt < 4; ++nt) { // z = x @ enc + b; delta
    const int c = nt * 16 + mn;
    v8f acc; const float bi = enc_b[g * DPD + c];
#pragma unroll
    for (int r = 0; r < 8; ++r) acc[r] = bi;
    acc = gemm16<DIM>(ax, DIM, encw + nt * 16 * DIM, DIM, lane, acc);
#pragma unroll
    for (int r = 0; r < 8; ++r) {
      const int row = half * 8 + r;
      const long long fr = (long long)(tt * 16 + row) * GRP + g;
      const float zv = acc[r];
      const float dv = zv - z_hat_prev[fr * DPD + c];
      out_z[fr * DPD + c] = zv;
      az[row * DPD + c]  = (bf16_t)zv;
      adz[row * DPD + c] = (bf16_t)dv;
      dfull[row * DPD + c] = dv;
    }
  }
  __syncthreads();

  if (lane < 16) { // surprise + gate
    const float* dr = dfull + lane * DPD;
    float s = 0.f;
#pragma unroll
    for (int i = 0; i < DPD; ++i) s += dr[i] * dr[i];
    const float sur = sqrtf(s);
    const long long fr = (long long)(tt * 16 + lane) * GRP + g;
    out_sur[fr]  = sur;
    out_gate[fr] = fminf(fmaxf(sur, 0.f), 1.f);
  }

  const bf16_t* gw = gain_t + (long long)g * DIM * DPD;
#pragma unroll 1
  for (int nt = 0; nt < 8; ++nt) { // gain = 1 + 0.1*tanh(delta @ gain_w + b)
    if (nt + 1 < 8) __builtin_prefetch(gw + (nt + 1) * 16 * DPD, 0, 1);
    const int c = nt * 16 + mn;
    v8f acc; const float bi = gain_b[g * DIM + c];
#pragma unroll
    for (int r = 0; r < 8; ++r) acc[r] = bi;
    acc = gemm16<DPD>(adz, DPD, gw + nt * 16 * DPD, DPD, lane, acc);
#pragma unroll
    for (int r = 0; r < 8; ++r)
      gains[(half * 8 + r) * DIM + c] = 1.f + 0.1f * tanhf(acc[r]);
  }

  const bf16_t* pw = pred_t + (long long)g * DPD * DPD;
#pragma unroll 1
  for (int nt = 0; nt < 4; ++nt) { // z_hat = z @ pred + b
    const int c = nt * 16 + mn;
    v8f acc; const float bi = pred_b[g * DPD + c];
#pragma unroll
    for (int r = 0; r < 8; ++r) acc[r] = bi;
    acc = gemm16<DPD>(az, DPD, pw + nt * 16 * DPD, DPD, lane, acc);
#pragma unroll
    for (int r = 0; r < 8; ++r) {
      const int row = half * 8 + r;
      const long long fr = (long long)(tt * 16 + row) * GRP + g;
      out_zhat[fr * DPD + c] = acc[r];
    }
  }
  __syncthreads();

  { // grouped LN * gain -> ah
    const int row = lane >> 1, seg = lane & 1;
    const float* xr = xres + row * DIM + seg * 64;
    float s = 0.f, s2 = 0.f;
#pragma unroll
    for (int t = 0; t < 64; ++t) { float v = xr[t]; s += v; s2 += v * v; }
    red[lane] = s; red[32 + lane] = s2;
    __syncthreads();
    const float sum = red[row * 2] + red[row * 2 + 1];
    const float sq  = red[32 + row * 2] + red[32 + row * 2 + 1];
    const float mean = sum * (1.f / DIM);
    const float var  = sq * (1.f / DIM) - mean * mean;
    const float rstd = rsqrtf(var + 1e-5f);
    const float* gg = ln_g + g * DIM + seg * 64;
    const float* bb = ln_b + g * DIM + seg * 64;
    const float* gn = gains + row * DIM + seg * 64;
    bf16_t* hr = ah + row * DIM + seg * 64;
#pragma unroll
    for (int t = 0; t < 64; ++t)
      hr[t] = (bf16_t)(((xr[t] - mean) * rstd * gg[t] + bb[t]) * gn[t]);
  }
  __syncthreads();

  const bf16_t* uw = up_t + (long long)g * HID * DIM;
#pragma unroll 1
  for (int nt = 0; nt < 32; ++nt) { // up + gelu(exact)
    if (nt + 1 < 32) __builtin_prefetch(uw + (nt + 1) * 16 * DIM, 0, 1);
    const int c = nt * 16 + mn;
    v8f acc; const float bi = up_b[g * HID + c];
#pragma unroll
    for (int r = 0; r < 8; ++r) acc[r] = bi;
    acc = gemm16<DIM>(ah, DIM, uw + nt * 16 * DIM, DIM, lane, acc);
#pragma unroll
    for (int r = 0; r < 8; ++r) {
      const float v = acc[r];
      const float ge = 0.5f * v * (1.f + erff(v * 0.70710678118654752f));
      au[(half * 8 + r) * HID + c] = (bf16_t)ge;
    }
  }
  __syncthreads();

  const bf16_t* dw = down_t + (long long)g * DIM * HID;
#pragma unroll 1
  for (int nt = 0; nt < 8; ++nt) { // down + residual
    if (nt + 1 < 8) __builtin_prefetch(dw + (nt + 1) * 16 * HID, 0, 1);
    const int c = nt * 16 + mn;
    v8f acc; const float bi = down_b[g * DIM + c];
#pragma unroll
    for (int r = 0; r < 8; ++r) acc[r] = bi;
    acc = gemm16<HID>(au, HID, dw + nt * 16 * HID, HID, lane, acc);
#pragma unroll
    for (int r = 0; r < 8; ++r) {
      const int row = half * 8 + r;
      const long long fr = (long long)(tt * 16 + row) * GRP + g;
      out_x[fr * DIM + c] = xres[row * DIM + c] + acc[r];
    }
  }
}

// ---------------------------------------------------------------------------
// Stage 3: post projection (128 -> 513) + normalize/sigmoid/split.
__global__ __launch_bounds__(32)
void k_post(const float* __restrict__ x_out,
            const bf16_t* __restrict__ post_t, const float* __restrict__ post_b,
            float* __restrict__ out_kcand, float* __restrict__ out_vpost,
            float* __restrict__ out_qnov, float* __restrict__ out_vcand,
            float* __restrict__ out_wnov) {
  const int NP = 4 * DIM + 1; // 513
  __shared__ float  xsf[16 * DIM];
  __shared__ bf16_t ax[16 * DIM];
  __shared__ float proj[16 * 513];
  __shared__ float inv1[16], inv2[16];

  const int lane = threadIdx.x;
  const int g  = blockIdx.x >> 7;
  const int tt = blockIdx.x & 127;
  const int half = lane >> 4, mn = lane & 15;

#pragma unroll
  for (int c = 0; c < 16; ++c) {
    const int chunk = c * 32 + lane;
    const int row = chunk >> 5, col4 = chunk & 31;
    const long long fr = (long long)(tt * 16 + row) * GRP + g;
    async_copy16(xsf + chunk * 4, x_out + fr * DIM + col4 * 4);
  }
  async_wait();
  __syncthreads();
  for (int i = lane; i < 16 * DIM; i += 32) ax[i] = (bf16_t)xsf[i];
  __syncthreads();

  const bf16_t* pw = post_t + (long long)g * NP * DIM;
#pragma unroll 1
  for (int nt = 0; nt < 32; ++nt) {
    if (nt + 1 < 32) __builtin_prefetch(pw + (nt + 1) * 16 * DIM, 0, 1);
    const int c = nt * 16 + mn;
    v8f acc; const float bi = post_b[g * NP + c];
#pragma unroll
    for (int r = 0; r < 8; ++r) acc[r] = bi;
    acc = gemm16<DIM>(ax, DIM, pw + nt * 16 * DIM, DIM, lane, acc);
#pragma unroll
    for (int r = 0; r < 8; ++r)
      proj[(half * 8 + r) * NP + c] = acc[r];
  }
  __syncthreads();

  if (lane < 16) { // last column + per-row norms
    const bf16_t* xr = ax + lane * DIM;
    const bf16_t* wr = pw + (long long)512 * DIM;
    float s = post_b[g * NP + 512];
#pragma unroll
    for (int i = 0; i < DIM; ++i) s += (float)xr[i] * (float)wr[i];
    const long long fr = (long long)(tt * 16 + lane) * GRP + g;
    out_wnov[fr] = 1.f / (1.f + __expf(-s));
    const float* pr = proj + lane * NP;
    float n1 = 0.f, n2 = 0.f;
#pragma unroll
    for (int i = 0; i < DIM; ++i) { n1 += pr[i] * pr[i]; n2 += pr[256 + i] * pr[256 + i]; }
    inv1[lane] = 1.f / fmaxf(sqrtf(n1), 1e-6f);
    inv2[lane] = 1.f / fmaxf(sqrtf(n2), 1e-6f);
  }
  __syncthreads();

  for (int i = lane; i < 16 * DIM; i += 32) {
    const int row = i >> 7, col = i & 127;
    const long long fr = (long long)(tt * 16 + row) * GRP + g;
    const float* pr = proj + row * NP;
    out_kcand[fr * DIM + col] = pr[col] * inv1[row];
    out_vpost[fr * DIM + col] = pr[DIM + col];
    out_qnov [fr * DIM + col] = pr[2 * DIM + col] * inv2[row];
    out_vcand[fr * DIM + col] = pr[3 * DIM + col];
  }
}

// ---------------------------------------------------------------------------
extern "C" void kernel_launch(void* const* d_in, const int* in_sizes, int n_in,
                              void* d_out, int out_size, void* d_ws, size_t ws_size,
                              hipStream_t stream) {
  (void)in_sizes; (void)n_in; (void)out_size; (void)ws_size;
  const float* x_col  = (const float*)d_in[0];
  const float* zhatp  = (const float*)d_in[1];
  const float* ln_g   = (const float*)d_in[2];
  const float* ln_b   = (const float*)d_in[3];
  const float* up_w   = (const float*)d_in[4];
  const float* up_b   = (const float*)d_in[5];
  const float* down_w = (const float*)d_in[6];
  const float* down_b = (const float*)d_in[7];
  const float* lat_g  = (const float*)d_in[8];
  const float* lat_b  = (const float*)d_in[9];
  const float* wq     = (const float*)d_in[10];
  const float* bq     = (const float*)d_in[11];
  const float* wk     = (const float*)d_in[12];
  const float* bk     = (const float*)d_in[13];
  const float* wv     = (const float*)d_in[14];
  const float* bv     = (const float*)d_in[15];
  const float* wo     = (const float*)d_in[16];
  const float* bo     = (const float*)d_in[17];
  const float* post_w = (const float*)d_in[18];
  const float* post_b = (const float*)d_in[19];
  const float* enc_w  = (const float*)d_in[20];
  const float* enc_b  = (const float*)d_in[21];
  const float* pred_w = (const float*)d_in[22];
  const float* pred_b = (const float*)d_in[23];
  const float* gain_w = (const float*)d_in[24];
  const float* gain_b = (const float*)d_in[25];

  const long long R = (long long)BSZ * SEQ * GRP; // 131072 rows
  const int TT = (BSZ * SEQ) / 16;                // 128 token tiles

  // workspace carve-up
  char* ws = (char*)d_ws;
  size_t off = 0;
  auto carve = [&](size_t bytes) -> void* {
    void* p = ws + off;
    off = (off + bytes + 255) & ~(size_t)255;
    return p;
  };
  bf16_t* wq_t   = (bf16_t*)carve((size_t)DIM * DIM * 2);
  bf16_t* wk_t   = (bf16_t*)carve((size_t)DIM * DIM * 2);
  bf16_t* wv_t   = (bf16_t*)carve((size_t)DIM * DIM * 2);
  bf16_t* wo_t   = (bf16_t*)carve((size_t)DIM * DIM * 2);
  bf16_t* enc_t  = (bf16_t*)carve((size_t)GRP * DPD * DIM * 2);
  bf16_t* pred_t = (bf16_t*)carve((size_t)GRP * DPD * DPD * 2);
  bf16_t* gain_t = (bf16_t*)carve((size_t)GRP * DIM * DPD * 2);
  bf16_t* up_t   = (bf16_t*)carve((size_t)GRP * HID * DIM * 2);
  bf16_t* down_t = (bf16_t*)carve((size_t)GRP * DIM * HID * 2);
  bf16_t* post_t = (bf16_t*)carve((size_t)GRP * 513 * DIM * 2);
  float*  x_mid  = (float*)carve((size_t)R * DIM * 4);

  // d_out slices (tuple concatenated flat, fp32)
  float* out    = (float*)d_out;
  float* o_xout = out;
  float* o_z    = o_xout + R * DIM;
  float* o_zhat = o_z + R * DPD;
  float* o_sur  = o_zhat + R * DPD;
  float* o_kc   = o_sur + R;
  float* o_vp   = o_kc + R * DIM;
  float* o_gate = o_vp + R * DIM;
  float* o_qn   = o_gate + R;
  float* o_vc   = o_qn + R * DIM;
  float* o_wn   = o_vc + R * DIM;

  auto tlaunch = [&](const float* s, bf16_t* d, int G, int K, int N) {
    long long tot = (long long)G * K * N;
    int blocks = (int)((tot + 255) / 256);
    k_transpose_bf16<<<blocks, 256, 0, stream>>>(s, d, G, K, N);
  };
  tlaunch(wq, wq_t, 1, DIM, DIM);
  tlaunch(wk, wk_t, 1, DIM, DIM);
  tlaunch(wv, wv_t, 1, DIM, DIM);
  tlaunch(wo, wo_t, 1, DIM, DIM);
  tlaunch(enc_w, enc_t, GRP, DIM, DPD);
  tlaunch(pred_w, pred_t, GRP, DPD, DPD);
  tlaunch(gain_w, gain_t, GRP, DPD, DIM);
  tlaunch(up_w, up_t, GRP, DIM, HID);
  tlaunch(down_w, down_t, GRP, HID, DIM);
  tlaunch(post_w, post_t, GRP, DIM, 513);

  k_attention<<<(int)(R / 16), 32, 0, stream>>>(
      x_col, lat_g, lat_b, wq_t, wk_t, wv_t, wo_t, bq, bk, bv, bo, x_mid);

  k_group_stage<<<GRP * TT, 32, 0, stream>>>(
      x_mid, zhatp, enc_t, enc_b, gain_t, gain_b, pred_t, pred_b,
      ln_g, ln_b, up_t, up_b, down_t, down_b,
      o_z, o_zhat, o_sur, o_gate, o_xout);

  k_post<<<GRP * TT, 32, 0, stream>>>(
      o_xout, post_t, post_b, o_kc, o_vp, o_qn, o_vc, o_wn);
}
